// GMLM_41867341201762
// MI455X (gfx1250) — matrix operand order repaired and checked by reading, not compiled, not run on hardware
//
#include <hip/hip_runtime.h>
#include <hip/hip_bf16.h>

typedef __bf16 bf16_t;
typedef __attribute__((ext_vector_type(16))) __bf16 v16bf;
typedef __attribute__((ext_vector_type(8)))  __bf16 v8bf;
typedef __attribute__((ext_vector_type(8)))  float  v8f;

#define CDIV(a, b) (((a) + (b) - 1) / (b))

// ---------------------------------------------------------------------------
// WMMA fragment loaders (contiguous k order k = 16h + e, wave32)
// ---------------------------------------------------------------------------
// A 16x32 bf16: lane l (m=l&15, h=l>>4) holds row m; elements 0..15 = K[h*16 .. h*16+15]
// -> two contiguous 16B loads.
__device__ inline v16bf load_a_frag(const bf16_t* __restrict__ A, int row, int ldA,
                                    int kb, int h) {
  const bf16_t* p0 = A + (size_t)row * ldA + kb + h * 16;
  v8bf lo = *(const v8bf*)(p0);
  v8bf hi = *(const v8bf*)(p0 + 8);
  v16bf a;
#pragma unroll
  for (int e = 0; e < 8; ++e) { a[e] = lo[e]; a[e + 8] = hi[e]; }
  return a;
}

// B fragments are PRE-PACKED: Wp[kt][ct][lane][16] with lane = h*16 + n and
// element e = W[kt*32 + h*16 + e][ct*16 + n]. One contiguous 32B load per lane.
__device__ inline v16bf load_b_frag_packed(const bf16_t* __restrict__ Wp,
                                           int kt, int nct, int ct, int lane) {
  const bf16_t* p = Wp + (((size_t)(kt * nct + ct) * 32) + lane) * 16;
  return *(const v16bf*)p;
}

// ---------------------------------------------------------------------------
// Generic fused WMMA GEMM:
//   acc1 = A[N,Kp] @ W[Kp,ncols]          (bf16 x bf16 -> f32, W packed)
//   acc2 = A2[N,Kp2] @ W2[Kp2,ncols]      (optional second GEMM, e.g. skip path)
//   out  = (relu? max(acc1+bias,0) : acc1+bias) + (acc2+bias2) + addbuf
// Block = 128 threads = 4 waves; each wave owns one 16x16 output tile.
// ---------------------------------------------------------------------------
__global__ void wmma_gemm_kernel(const bf16_t* __restrict__ A, int Kp,
                                 const bf16_t* __restrict__ W,
                                 const float* __restrict__ bias,
                                 const bf16_t* __restrict__ A2, int Kp2,
                                 const bf16_t* __restrict__ W2,
                                 const float* __restrict__ bias2,
                                 const float* __restrict__ addbuf, int ld_add,
                                 int do_relu,
                                 float* __restrict__ out_f, int ldo_f,
                                 bf16_t* __restrict__ out_bf, int ldo_bf,
                                 int n_row_tiles, int ncols) {
  const int wave = threadIdx.x >> 5;
  const int lane = threadIdx.x & 31;
  const int rowTile = blockIdx.y * 4 + wave;
  if (rowTile >= n_row_tiles) return;  // wave-uniform: EXEC stays all-ones for WMMA
  const int ct = blockIdx.x;
  const int c0 = ct * 16;
  const int nct = ncols >> 4;
  const int h = lane >> 4;
  const int m = lane & 15;
  const int row0 = rowTile * 16;

  v8f acc;
#pragma unroll
  for (int i = 0; i < 8; ++i) acc[i] = 0.0f;

  const int nkt = Kp >> 5;
  for (int kt = 0; kt < nkt; ++kt) {
    v16bf a = load_a_frag(A, row0 + m, Kp, kt * 32, h);
    v16bf b = load_b_frag_packed(W, kt, nct, ct, lane);
    acc = __builtin_amdgcn_wmma_f32_16x16x32_bf16(false, a, false, b,
                                                  (short)0, acc, false, false);
  }

  v8f acc2;
#pragma unroll
  for (int i = 0; i < 8; ++i) acc2[i] = 0.0f;
  if (A2) {
    const int nkt2 = Kp2 >> 5;
    for (int kt = 0; kt < nkt2; ++kt) {
      v16bf a = load_a_frag(A2, row0 + m, Kp2, kt * 32, h);
      v16bf b = load_b_frag_packed(W2, kt, nct, ct, lane);
      acc2 = __builtin_amdgcn_wmma_f32_16x16x32_bf16(false, a, false, b,
                                                     (short)0, acc2, false, false);
    }
  }

  const float bv  = bias  ? bias[c0 + m]  : 0.0f;
  const float bv2 = (A2 && bias2) ? bias2[c0 + m] : 0.0f;
  const int rbase = row0 + h * 8;  // D layout: VGPR r -> row r (+8 for upper lanes)
#pragma unroll
  for (int r = 0; r < 8; ++r) {
    float v = acc[r] + bv;
    if (do_relu) v = fmaxf(v, 0.0f);
    if (A2) v += acc2[r] + bv2;
    const int orow = rbase + r;
    const int ocol = c0 + m;
    if (addbuf) v += addbuf[(size_t)orow * ld_add + ocol];
    if (out_f)  out_f[(size_t)orow * ldo_f + ocol] = v;
    if (out_bf) out_bf[(size_t)orow * ldo_bf + ocol] = (bf16_t)v;
  }
}

// ---------------------------------------------------------------------------
// Elementwise / scatter helpers
// ---------------------------------------------------------------------------
__global__ void zero_kernel(float4* __restrict__ p, int n4) {
  int i = blockIdx.x * blockDim.x + threadIdx.x;
  int stride = gridDim.x * blockDim.x;
  for (; i < n4; i += stride) p[i] = make_float4(0.f, 0.f, 0.f, 0.f);
}

__global__ void softmax16_kernel(const float* __restrict__ rw, float* __restrict__ cw) {
  if (threadIdx.x == 0 && blockIdx.x == 0) {
    float mx = rw[0];
    for (int i = 1; i < 16; ++i) mx = fmaxf(mx, rw[i]);
    float e[16], s = 0.f;
    for (int i = 0; i < 16; ++i) { e[i] = expf(rw[i] - mx); s += e[i]; }
    for (int i = 0; i < 16; ++i) cw[i] = e[i] / s;
  }
}

__global__ void degree_kernel(const int* __restrict__ src, const int* __restrict__ dst,
                              float* __restrict__ deg_out, float* __restrict__ deg_in, int E) {
  int e = blockIdx.x * blockDim.x + threadIdx.x;
  if (e < E) {
    atomicAdd(&deg_out[src[e]], 1.0f);
    atomicAdd(&deg_in[dst[e]], 1.0f);
  }
}

__global__ void gcn_norms_kernel(const float* __restrict__ deg_out,
                                 const float* __restrict__ deg_in,
                                 float* __restrict__ ns, float* __restrict__ nd, int n) {
  int i = blockIdx.x * blockDim.x + threadIdx.x;
  if (i < n) {
    ns[i] = (deg_out[i] > 0.f) ? rsqrtf(fmaxf(deg_out[i], 1.0f)) : 0.0f;
    nd[i] = (deg_in[i]  > 0.f) ? rsqrtf(fmaxf(deg_in[i],  1.0f)) : 0.0f;
  }
}

// m1 += (x*cw)[src] at dst, 16 feats/edge
__global__ void scatter_m1_kernel(const int* __restrict__ src, const int* __restrict__ dst,
                                  const float* __restrict__ x, const float* __restrict__ cw,
                                  float* __restrict__ m1, int E) {
  int t = blockIdx.x * blockDim.x + threadIdx.x;
  if (t >= E * 16) return;
  int e = t >> 4, f = t & 15;
  atomicAdd(&m1[(size_t)dst[e] * 16 + f], x[(size_t)src[e] * 16 + f] * cw[f]);
}

// m2 += out1[src] at dst, 256 feats/edge, float4 granularity
__global__ void scatter_m2_kernel(const int* __restrict__ src, const int* __restrict__ dst,
                                  const float* __restrict__ out1, float* __restrict__ m2, int E) {
  int t = blockIdx.x * blockDim.x + threadIdx.x;
  if (t >= E * 64) return;
  int e = t >> 6, q = (t & 63) << 2;
  float4 v = *(const float4*)(out1 + (size_t)src[e] * 256 + q);
  float* d = m2 + (size_t)dst[e] * 256 + q;
  atomicAdd(d + 0, v.x); atomicAdd(d + 1, v.y);
  atomicAdd(d + 2, v.z); atomicAdd(d + 3, v.w);
}

// generic F-wide scatter-add (GCN aggregations)
__global__ void scatter_generic_kernel(const int* __restrict__ src, const int* __restrict__ dst,
                                       const float* __restrict__ feat, float* __restrict__ acc,
                                       int E, int F) {
  int t = blockIdx.x * blockDim.x + threadIdx.x;
  if (t >= E * F) return;
  int e = t / F, f = t - e * F;
  atomicAdd(&acc[(size_t)dst[e] * F + f], feat[(size_t)src[e] * F + f]);
}

// build bf16 A for layer1: [wx | m1/deg] packed [N,32]
__global__ void finalize_l1_inputs_kernel(const float* __restrict__ x, const float* __restrict__ cw,
                                          const float* __restrict__ m1, const float* __restrict__ deg_in,
                                          bf16_t* __restrict__ ab, int n) {
  int t = blockIdx.x * blockDim.x + threadIdx.x;
  if (t >= n * 16) return;
  int i = t >> 4, f = t & 15;
  float w = x[t] * cw[f];
  float mm = m1[t] / fmaxf(deg_in[i], 1.0f);
  ab[(size_t)i * 32 + f] = (bf16_t)w;
  ab[(size_t)i * 32 + 16 + f] = (bf16_t)mm;
}

// m2/deg -> second half of cat2 bf16 [N,512]
__global__ void finalize_m2_kernel(const float* __restrict__ m2, const float* __restrict__ deg_in,
                                   bf16_t* __restrict__ cat2, int n) {
  int t = blockIdx.x * blockDim.x + threadIdx.x;
  if (t >= n * 256) return;
  int i = t >> 8, j = t & 255;
  cat2[(size_t)i * 512 + 256 + j] = (bf16_t)(m2[t] / fmaxf(deg_in[i], 1.0f));
}

// out[i,f] = feat[i,f] * s[i]
__global__ void scale_rows_kernel(const float* __restrict__ feat, const float* __restrict__ s,
                                  float* __restrict__ out, int n, int F) {
  int t = blockIdx.x * blockDim.x + threadIdx.x;
  if (t >= n * F) return;
  int i = t / F;
  out[t] = feat[t] * s[i];
}

// bf16 A for GCN matmul: A[i,k] = (k<F) ? agg[i,k]*nd[i] : 0   (pad K to Kp)
__global__ void gcn_a_finalize_kernel(const float* __restrict__ agg, const float* __restrict__ nd,
                                      bf16_t* __restrict__ out, int n, int F, int Kp) {
  int t = blockIdx.x * blockDim.x + threadIdx.x;
  if (t >= n * Kp) return;
  int i = t / Kp, k = t - i * Kp;
  float v = (k < F) ? agg[(size_t)i * F + k] * nd[i] : 0.0f;
  out[t] = (bf16_t)v;
}

__global__ void score_kernel(const float* __restrict__ emb, const float* __restrict__ Wfc,
                             const float* __restrict__ bfc, float* __restrict__ scores, int n) {
  int i = blockIdx.x * blockDim.x + threadIdx.x;
  if (i >= n) return;
  float s = bfc[0];
  const float* row = emb + (size_t)i * 256;
#pragma unroll 8
  for (int j = 0; j < 256; ++j) s += row[j] * Wfc[j];
  scores[i] = s;
}

__global__ void primary_kernel(const int* __restrict__ pairs, const float* __restrict__ scores,
                               float* __restrict__ out, int P) {
  int p = blockIdx.x * blockDim.x + threadIdx.x;
  if (p >= P) return;
  float d = scores[pairs[2 * p]] - scores[pairs[2 * p + 1]];
  out[p] = 1.0f / (1.0f + expf(-d));
}

__global__ void attn_kernel(const float* __restrict__ hid, const float* __restrict__ Wa2,
                            const float* __restrict__ ba2, float* __restrict__ out, int n) {
  int i = blockIdx.x * blockDim.x + threadIdx.x;
  if (i >= n) return;
  float s = ba2[0];
  const float* row = hid + (size_t)i * 128;
#pragma unroll 8
  for (int j = 0; j < 128; ++j) s += row[j] * Wa2[j];
  out[i] = 1.0f / (1.0f + expf(-s));
}

// ---------------------------------------------------------------------------
// Pack all weights to bf16 in B-FRAGMENT layout: Wp[kt][ct][lane][e] with
//   k = kt*32 + (lane>>4)*16 + e,  c = ct*16 + (lane&15)
// so a wave's B fragment is one contiguous 32B load per lane.
// Arena (elements): [0,8192) Wcat=[Ws1;Wn1] K32xC256 | [8192,16384) WskipP K32xC256
// [16384,147456) Wcat2=[Ws2;Wn2] K512xC256 | [147456,148992) Wg1P K32xC48
// [148992,165376) Wg2P K64xC256 | [165376,198144) Wa1 K256xC128
// ---------------------------------------------------------------------------
__device__ inline void unpack_frag_idx(int u, int nct, int& k, int& c) {
  int e = u & 15;
  int lane = (u >> 4) & 31;
  int frag = u >> 9;
  int kt = frag / nct, ct = frag - kt * nct;
  k = kt * 32 + (lane >> 4) * 16 + e;
  c = ct * 16 + (lane & 15);
}

__global__ void pack_weights_kernel(const float* __restrict__ Ws1, const float* __restrict__ Wn1,
                                    const float* __restrict__ Wskip,
                                    const float* __restrict__ Ws2, const float* __restrict__ Wn2,
                                    const float* __restrict__ Wg1, const float* __restrict__ Wg2,
                                    const float* __restrict__ Wa1, bf16_t* __restrict__ wb) {
  int t = blockIdx.x * blockDim.x + threadIdx.x;
  if (t >= 198144) return;
  float v;
  int k, c;
  if (t < 8192) {                       // Wcat: rows 0-15 Ws1, 16-31 Wn1
    unpack_frag_idx(t, 16, k, c);
    v = (k < 16) ? Ws1[k * 256 + c] : Wn1[(k - 16) * 256 + c];
  } else if (t < 16384) {               // WskipP: rows >=16 zero
    unpack_frag_idx(t - 8192, 16, k, c);
    v = (k < 16) ? Wskip[k * 256 + c] : 0.0f;
  } else if (t < 147456) {              // Wcat2: rows 0-255 Ws2, 256-511 Wn2
    unpack_frag_idx(t - 16384, 16, k, c);
    v = (k < 256) ? Ws2[k * 256 + c] : Wn2[(k - 256) * 256 + c];
  } else if (t < 148992) {              // Wg1P: K pad 16->32
    unpack_frag_idx(t - 147456, 3, k, c);
    v = (k < 16) ? Wg1[k * 48 + c] : 0.0f;
  } else if (t < 165376) {              // Wg2P: K pad 48->64
    unpack_frag_idx(t - 148992, 16, k, c);
    v = (k < 48) ? Wg2[k * 256 + c] : 0.0f;
  } else {                              // Wa1P
    unpack_frag_idx(t - 165376, 8, k, c);
    v = Wa1[k * 128 + c];
  }
  wb[t] = (bf16_t)v;
}

// ---------------------------------------------------------------------------
extern "C" void kernel_launch(void* const* d_in, const int* in_sizes, int n_in,
                              void* d_out, int out_size, void* d_ws, size_t ws_size,
                              hipStream_t stream) {
  const float* x   = (const float*)d_in[0];
  const int*   src = (const int*)d_in[1];
  const int*   dst = (const int*)d_in[2];
  const int*   pairs = (const int*)d_in[3];
  const float* raw_w = (const float*)d_in[4];
  const float* Ws1 = (const float*)d_in[5];
  const float* Wn1 = (const float*)d_in[6];
  const float* b1  = (const float*)d_in[7];
  const float* Ws2 = (const float*)d_in[8];
  const float* Wn2 = (const float*)d_in[9];
  const float* b2  = (const float*)d_in[10];
  const float* Wskip = (const float*)d_in[11];
  const float* bskip = (const float*)d_in[12];
  const float* Wg1 = (const float*)d_in[13];
  const float* bg1 = (const float*)d_in[14];
  const float* Wg2 = (const float*)d_in[15];
  const float* bg2 = (const float*)d_in[16];
  const float* Wfc = (const float*)d_in[17];
  const float* bfc = (const float*)d_in[18];
  // Wa1 = d_in[19] (packed)
  const float* ba1 = (const float*)d_in[20];
  const float* Wa2 = (const float*)d_in[21];
  const float* ba2 = (const float*)d_in[22];

  const int n = in_sizes[0] / 16;   // 100000 (multiple of 16)
  const int E = in_sizes[1];        // 1600000
  const int P = in_sizes[3] / 2;    // 100000
  float* out = (float*)d_out;       // [primary n | aux n*256 | attn n]

  // ---- workspace layout ----
  char* ws = (char*)d_ws;
  size_t off = 0;
  auto alloc = [&](size_t bytes) -> size_t {
    size_t o = off;
    off = (off + bytes + 255) & ~(size_t)255;
    return o;
  };
  // zeroed accumulation arena (contiguous)
  size_t zero_base = off;
  size_t o_deg_in  = alloc((size_t)n * 4);
  size_t o_deg_out = alloc((size_t)n * 4);
  size_t o_m1      = alloc((size_t)n * 16 * 4);
  size_t o_m2      = alloc((size_t)n * 256 * 4);
  size_t o_agg1    = alloc((size_t)n * 16 * 4);
  size_t o_agg2    = alloc((size_t)n * 48 * 4);
  size_t zero_end = off;
  // non-zeroed
  size_t o_cw     = alloc(64);
  size_t o_ns     = alloc((size_t)n * 4);
  size_t o_nd     = alloc((size_t)n * 4);
  size_t o_ab1    = alloc((size_t)n * 32 * 2);   // [wx|m1] bf16
  size_t o_cat2   = alloc((size_t)n * 512 * 2);  // [out1|m2] bf16
  size_t o_out1   = alloc((size_t)n * 256 * 4);
  size_t o_emb    = alloc((size_t)n * 256 * 4);
  size_t o_embbf  = alloc((size_t)n * 256 * 2);
  size_t o_hid    = alloc((size_t)n * 128 * 4);
  size_t o_a1     = alloc((size_t)n * 48 * 4);
  size_t o_h      = alloc((size_t)n * 48 * 4);
  size_t o_g1a    = alloc((size_t)n * 32 * 2);
  size_t o_g2a    = alloc((size_t)n * 64 * 2);
  size_t o_scores = alloc((size_t)n * 4);
  size_t o_wb     = alloc((size_t)198144 * 2);
  (void)ws_size;

  float*  deg_in  = (float*)(ws + o_deg_in);
  float*  deg_out = (float*)(ws + o_deg_out);
  float*  m1      = (float*)(ws + o_m1);
  float*  m2      = (float*)(ws + o_m2);
  float*  agg1    = (float*)(ws + o_agg1);
  float*  agg2    = (float*)(ws + o_agg2);
  float*  cw      = (float*)(ws + o_cw);
  float*  ns      = (float*)(ws + o_ns);
  float*  nd      = (float*)(ws + o_nd);
  bf16_t* ab1     = (bf16_t*)(ws + o_ab1);
  bf16_t* cat2    = (bf16_t*)(ws + o_cat2);
  float*  out1    = (float*)(ws + o_out1);
  float*  emb     = (float*)(ws + o_emb);
  bf16_t* embbf   = (bf16_t*)(ws + o_embbf);
  float*  hid     = (float*)(ws + o_hid);
  float*  a1      = (float*)(ws + o_a1);
  float*  hbuf    = (float*)(ws + o_h);
  bf16_t* g1a     = (bf16_t*)(ws + o_g1a);
  bf16_t* g2a     = (bf16_t*)(ws + o_g2a);
  float*  scores  = (float*)(ws + o_scores);
  bf16_t* wb      = (bf16_t*)(ws + o_wb);

  bf16_t* Wcat   = wb;
  bf16_t* WskipP = wb + 8192;
  bf16_t* Wcat2  = wb + 16384;
  bf16_t* Wg1P   = wb + 147456;
  bf16_t* Wg2P   = wb + 148992;
  bf16_t* Wa1P   = wb + 165376;

  const int nrt = n / 16;  // row tiles
  const dim3 blk(256);
  const dim3 gblk(128);    // 4 waves per GEMM block

  // 0) zero accumulators (replay-safe)
  {
    int n4 = (int)((zero_end - zero_base) / 16);
    zero_kernel<<<dim3(2048), blk, 0, stream>>>((float4*)(ws + zero_base), n4);
  }
  // 1) softmax weights, packed bf16 weights
  softmax16_kernel<<<dim3(1), dim3(32), 0, stream>>>(raw_w, cw);
  pack_weights_kernel<<<dim3(CDIV(198144, 256)), blk, 0, stream>>>(
      Ws1, Wn1, Wskip, Ws2, Wn2, Wg1, Wg2, (const float*)d_in[19], wb);
  // 2) degrees and GCN norms
  degree_kernel<<<dim3(CDIV(E, 256)), blk, 0, stream>>>(src, dst, deg_out, deg_in, E);
  gcn_norms_kernel<<<dim3(CDIV(n, 256)), blk, 0, stream>>>(deg_out, deg_in, ns, nd, n);
  // 3) m1 scatter and layer-1 A assembly
  scatter_m1_kernel<<<dim3(CDIV(E * 16, 256)), blk, 0, stream>>>(src, dst, x, cw, m1, E);
  finalize_l1_inputs_kernel<<<dim3(CDIV(n * 16, 256)), blk, 0, stream>>>(x, cw, m1, deg_in, ab1, n);
  // 4) layer1: out1 = relu([wx|m1]@[Ws1;Wn1] + b1) + wx@Wskip + bskip
  wmma_gemm_kernel<<<dim3(256 / 16, CDIV(nrt, 4)), gblk, 0, stream>>>(
      ab1, 32, Wcat, b1, ab1, 32, WskipP, bskip,
      nullptr, 0, 1, out1, 256, cat2, 512, nrt, 256);
  // 5) m2 scatter + finalize into cat2[:,256:512]
  scatter_m2_kernel<<<dim3(CDIV(E * 64, 256)), blk, 0, stream>>>(src, dst, out1, m2, E);
  finalize_m2_kernel<<<dim3(CDIV(n * 256, 256)), blk, 0, stream>>>(m2, deg_in, cat2, n);
  // 6) layer2: emb = relu([out1|m2]@[Ws2;Wn2] + b2) + out1
  wmma_gemm_kernel<<<dim3(256 / 16, CDIV(nrt, 4)), gblk, 0, stream>>>(
      cat2, 512, Wcat2, b2, nullptr, 0, nullptr, nullptr,
      out1, 256, 1, emb, 256, embbf, 256, nrt, 256);
  // 7) scores -> primary
  score_kernel<<<dim3(CDIV(n, 256)), blk, 0, stream>>>(emb, Wfc, bfc, scores, n);
  primary_kernel<<<dim3(CDIV(P, 256)), blk, 0, stream>>>(pairs, scores, out, P);
  // 8) GCN layer 1: a1 = relu(((x*ns) agg * nd) @ Wg1 + bg1)
  scale_rows_kernel<<<dim3(CDIV(n * 16, 256)), blk, 0, stream>>>(x, ns, hbuf, n, 16);
  scatter_generic_kernel<<<dim3(CDIV(E * 16, 256)), blk, 0, stream>>>(src, dst, hbuf, agg1, E, 16);
  gcn_a_finalize_kernel<<<dim3(CDIV(n * 32, 256)), blk, 0, stream>>>(agg1, nd, g1a, n, 16, 32);
  wmma_gemm_kernel<<<dim3(48 / 16, CDIV(nrt, 4)), gblk, 0, stream>>>(
      g1a, 32, Wg1P, bg1, nullptr, 0, nullptr, nullptr,
      nullptr, 0, 1, a1, 48, nullptr, 0, nrt, 48);
  // 9) GCN layer 2 -> aux output (d_out + n, [n,256])
  scale_rows_kernel<<<dim3(CDIV(n * 48, 256)), blk, 0, stream>>>(a1, ns, hbuf, n, 48);
  scatter_generic_kernel<<<dim3(CDIV(E * 48, 256)), blk, 0, stream>>>(src, dst, hbuf, agg2, E, 48);
  gcn_a_finalize_kernel<<<dim3(CDIV(n * 64, 256)), blk, 0, stream>>>(agg2, nd, g2a, n, 48, 64);
  wmma_gemm_kernel<<<dim3(256 / 16, CDIV(nrt, 4)), gblk, 0, stream>>>(
      g2a, 64, Wg2P, bg2, nullptr, 0, nullptr, nullptr,
      nullptr, 0, 0, out + n, 256, nullptr, 0, nrt, 256);
  // 10) attention head: hid = relu(emb@Wa1 + ba1); attn = sigmoid(hid@Wa2 + ba2)
  wmma_gemm_kernel<<<dim3(128 / 16, CDIV(nrt, 4)), gblk, 0, stream>>>(
      embbf, 256, Wa1P, ba1, nullptr, 0, nullptr, nullptr,
      nullptr, 0, 1, hid, 128, nullptr, 0, nrt, 128);
  attn_kernel<<<dim3(CDIV(n, 256)), blk, 0, stream>>>(hid, Wa2, ba2, out + (size_t)n * 257, n);
}
